// UniDrectionTransformer_51359218925812
// MI455X (gfx1250) — compile-verified, matrix operand order checked by reading, not run on hardware
//
#include <hip/hip_runtime.h>
#include <math.h>
#include <stdint.h>

// ---------------------------------------------------------------------------
// CDNA5 WMMA types (wave32): D(f32 16x16) = A(bf16 16x32) x B(bf16 32x16) + C
// ---------------------------------------------------------------------------
typedef __attribute__((ext_vector_type(16))) __bf16 bf16x16;
typedef __attribute__((ext_vector_type(8)))  __bf16 bf16x8;
typedef __attribute__((ext_vector_type(8)))  float  floatx8;
typedef __attribute__((ext_vector_type(4)))  int    intx4;

union BF16x16 { bf16x16 v; bf16x8 h[2]; };

__device__ __forceinline__ floatx8 wmma_bf16(bf16x16 a, bf16x16 b, floatx8 c) {
  return __builtin_amdgcn_wmma_f32_16x16x32_bf16(false, a, false, b, (short)0, c, false, false);
}
__device__ __forceinline__ bf16x8 cvt8(floatx8 f) {
  return __builtin_convertvector(f, bf16x8);
}

// ---- async global->LDS (gfx1250), guarded so compile can never break ------
#if defined(__has_builtin)
#if __has_builtin(__builtin_amdgcn_global_load_async_to_lds_b128) && \
    __has_builtin(__builtin_amdgcn_s_wait_asynccnt)
#define USE_ASYNC_LDS 1
#endif
#endif

#define AS1 __attribute__((address_space(1)))
#define AS3 __attribute__((address_space(3)))
__device__ __forceinline__ AS3 intx4* lds_i4(const void* p) {
  return (AS3 intx4*)(uint32_t)(uintptr_t)p;   // low 32 bits of generic LDS addr = LDS offset
}
__device__ __forceinline__ AS1 intx4* gbl_i4(const void* p) {
  return (AS1 intx4*)(uintptr_t)p;
}

// ---------------------------------------------------------------------------
// Weight transpose + convert: W[K,N] f32 -> Wt[N,K] bf16.  K,N multiples of 32.
// ---------------------------------------------------------------------------
__global__ __launch_bounds__(256) void k_wtrans(const float* __restrict__ W,
                                                __bf16* __restrict__ Wt,
                                                int K, int N) {
  __shared__ float tile[32][33];
  int n0 = blockIdx.x * 32, k0 = blockIdx.y * 32;
  int tx = threadIdx.x & 31, ty = threadIdx.x >> 5;   // 32 x 8
#pragma unroll
  for (int j = 0; j < 32; j += 8)
    tile[ty + j][tx] = W[(size_t)(k0 + ty + j) * N + n0 + tx];
  __syncthreads();
#pragma unroll
  for (int j = 0; j < 32; j += 8)
    Wt[(size_t)(n0 + ty + j) * K + k0 + tx] = (__bf16)tile[tx][ty + j];
}

// ---------------------------------------------------------------------------
// V transpose (f32): V[B,Sv,1024] -> VT[B,1024,SvP], zero-padded to SvP.
// ---------------------------------------------------------------------------
__global__ __launch_bounds__(256) void k_vtrans(const float* __restrict__ V,
                                                float* __restrict__ VT,
                                                int Sv, int SvP) {
  __shared__ float tile[32][33];
  int s0 = blockIdx.x * 32, d0 = blockIdx.y * 32, b = blockIdx.z;
  int tx = threadIdx.x & 31, ty = threadIdx.x >> 5;
#pragma unroll
  for (int j = 0; j < 32; j += 8) {
    int s = s0 + ty + j;
    tile[ty + j][tx] = (s < Sv) ? V[((size_t)b * Sv + s) * 1024 + d0 + tx] : 0.f;
  }
  __syncthreads();
#pragma unroll
  for (int j = 0; j < 32; j += 8)
    VT[((size_t)b * 1024 + d0 + ty + j) * SvP + s0 + tx] = tile[tx][ty + j];
}

// ---------------------------------------------------------------------------
// LayerNorm over last dim D, fp32. One 256-thread block per row.
// ---------------------------------------------------------------------------
__global__ __launch_bounds__(256) void k_layernorm(const float* __restrict__ x,
                                                   const float* __restrict__ g,
                                                   const float* __restrict__ b,
                                                   float* __restrict__ y, int D) {
  int row = blockIdx.x;
  const float* xr = x + (size_t)row * D;
  float* yr = y + (size_t)row * D;
  float s = 0.f, s2 = 0.f;
  for (int i = threadIdx.x; i < D; i += 256) { float v = xr[i]; s += v; s2 += v * v; }
  __shared__ float rs[256], rs2[256];
  rs[threadIdx.x] = s; rs2[threadIdx.x] = s2;
  __syncthreads();
  for (int off = 128; off > 0; off >>= 1) {
    if (threadIdx.x < off) { rs[threadIdx.x] += rs[threadIdx.x + off]; rs2[threadIdx.x] += rs2[threadIdx.x + off]; }
    __syncthreads();
  }
  float mu = rs[0] / (float)D;
  float var = rs2[0] / (float)D - mu * mu;
  float rstd = rsqrtf(var + 1e-5f);
  for (int i = threadIdx.x; i < D; i += 256)
    yr[i] = (xr[i] - mu) * rstd * g[i] + b[i];
}

// ---------------------------------------------------------------------------
// GEMM: C[M,N] f32 = A[M,K] f32 x Wt[N,K] bf16 (+bias[N]) (+resid[M,N])
// Block 256 thr (8 waves), tile 128x128x32; each wave: 2x4 16x16 accumulators
// -> 8 WMMAs per barrier pair.  N%128==0, K%32==0; M guarded.
// ---------------------------------------------------------------------------
#define BM 128
#define BN 128
#define BKK 32

__global__ __launch_bounds__(256) void k_gemm(const float* __restrict__ A,
                                              const __bf16* __restrict__ Wt,
                                              const float* __restrict__ bias,
                                              const float* __restrict__ resid,
                                              float* __restrict__ C,
                                              int M, int N, int K) {
  __shared__ __bf16 sA[BM][BKK];
  __shared__ __bf16 sB[BN][BKK];

  int tid = threadIdx.x, wave = tid >> 5, lane = tid & 31;
  int bm = blockIdx.y * BM, bn = blockIdx.x * BN;
  int wr = (wave & 3) * 32;     // wave row base (2 tiles)
  int wc = (wave >> 2) * 64;    // wave col base (4 tiles)
  int lm = lane & 15, lk = (lane >> 4) * 16;

  floatx8 acc[2][4] = {};

  int srow = tid >> 1;               // 128 rows
  int shalf = (tid & 1) * 16;        // 2 halves of 16

  int gr = bm + srow;
  int grc = gr < M ? gr : (M - 1);
  float mf = gr < M ? 1.f : 0.f;

  for (int k0 = 0; k0 < K; k0 += BKK) {
    // ---- stage A (f32 -> bf16), vectorized & branchless ----
    {
      const floatx8* ap = (const floatx8*)(A + (size_t)grc * K + k0 + shalf);
      floatx8 f0 = ap[0], f1 = ap[1];
      f0 *= mf; f1 *= mf;
      *(bf16x8*)&sA[srow][shalf]     = cvt8(f0);
      *(bf16x8*)&sA[srow][shalf + 8] = cvt8(f1);
      if (k0 + BKK < K) __builtin_prefetch((const float*)ap + BKK, 0, 1);
    }
    // ---- stage B (already bf16, N-major) ----
    {
      const __bf16* bp = Wt + (size_t)(bn + srow) * K + k0 + shalf;
#ifdef USE_ASYNC_LDS
      __builtin_amdgcn_global_load_async_to_lds_b128(gbl_i4(bp),     lds_i4(&sB[srow][shalf]),     0, 0);
      __builtin_amdgcn_global_load_async_to_lds_b128(gbl_i4(bp + 8), lds_i4(&sB[srow][shalf + 8]), 0, 0);
#else
      *(bf16x8*)&sB[srow][shalf]     = *(const bf16x8*)bp;
      *(bf16x8*)&sB[srow][shalf + 8] = *(const bf16x8*)(bp + 8);
#endif
    }
#ifdef USE_ASYNC_LDS
    __builtin_amdgcn_s_wait_asynccnt(0);
#endif
    __syncthreads();

    bf16x16 a0 = *(const bf16x16*)&sA[wr + lm][lk];
    bf16x16 a1 = *(const bf16x16*)&sA[wr + 16 + lm][lk];
    bf16x16 b0 = *(const bf16x16*)&sB[wc + lm][lk];
    bf16x16 b1 = *(const bf16x16*)&sB[wc + 16 + lm][lk];
    bf16x16 b2 = *(const bf16x16*)&sB[wc + 32 + lm][lk];
    bf16x16 b3 = *(const bf16x16*)&sB[wc + 48 + lm][lk];

    acc[0][0] = wmma_bf16(a0, b0, acc[0][0]);
    acc[0][1] = wmma_bf16(a0, b1, acc[0][1]);
    acc[0][2] = wmma_bf16(a0, b2, acc[0][2]);
    acc[0][3] = wmma_bf16(a0, b3, acc[0][3]);
    acc[1][0] = wmma_bf16(a1, b0, acc[1][0]);
    acc[1][1] = wmma_bf16(a1, b1, acc[1][1]);
    acc[1][2] = wmma_bf16(a1, b2, acc[1][2]);
    acc[1][3] = wmma_bf16(a1, b3, acc[1][3]);
    __syncthreads();
  }

  // ---- epilogue ----
  int mbase = bm + wr + (lane >> 4) * 8;
#pragma unroll
  for (int i = 0; i < 2; ++i) {
#pragma unroll
    for (int j = 0; j < 4; ++j) {
      int n = bn + wc + j * 16 + lm;
      float bv = bias ? bias[n] : 0.f;
#pragma unroll
      for (int r = 0; r < 8; ++r) {
        int m = mbase + i * 16 + r;
        if (m < M) {
          size_t idx = (size_t)m * N + n;
          float v = acc[i][j][r] + bv;
          if (resid) v += resid[idx];
          C[idx] = v;
        }
      }
    }
  }
}

// ---------------------------------------------------------------------------
// Attention, one 16-row q-tile per block, 128 threads (4 waves).
// Q,K: (B, S|Skv, 1024) f32, head h at cols [64h, 64h+64).  VT: (B,1024,SKP)
// f32, zero padded.  O: (B,S,1024) f32 (optional accumulate).
// LDS: scores[16][SKP] f32 | sQ[16][64] bf16 | red[16][8] | rowv[16]
// ---------------------------------------------------------------------------
__global__ __launch_bounds__(128) void k_attn16(const float* __restrict__ Q,
                                                const float* __restrict__ K,
                                                const float* __restrict__ VT,
                                                float* __restrict__ O,
                                                int Skv, float scale, int accum) {
  const int D = 1024, HD = 64;
  int qt = blockIdx.x, h = blockIdx.y, b = blockIdx.z;
  int S = gridDim.x * 16;
  int q0 = qt * 16;
  int SKP = (Skv + 31) & ~31;

  extern __shared__ char smem[];
  float*  sS   = (float*)smem;                               // 16*SKP
  __bf16* sQ   = (__bf16*)(smem + (size_t)16 * SKP * 4);     // 16*64
  float*  red  = (float*)(sQ + 16 * 64);                     // 16*8
  float*  rowv = red + 128;                                  // 16

  int tid = threadIdx.x, wave = tid >> 5, lane = tid & 31;
  int lm = lane & 15, lk = (lane >> 4) * 16;

  size_t qbase = ((size_t)b * S + q0) * D + h * HD;
  size_t kbase = ((size_t)b * Skv) * D + h * HD;

  // stage Q tile (16x64) -> bf16, one floatx8 per thread
  {
    int r = tid >> 3, c0 = (tid & 7) * 8;
    floatx8 f = *(const floatx8*)(Q + qbase + (size_t)r * D + c0);
    *(bf16x8*)&sQ[r * 64 + c0] = cvt8(f);
  }
  __syncthreads();

  bf16x16 aq0 = *(const bf16x16*)&sQ[lm * 64 + lk];
  bf16x16 aq1 = *(const bf16x16*)&sQ[lm * 64 + 32 + lk];

  // ---- scores: K^T fragments read contiguously from global ----
  int nkt = SKP >> 4;
  for (int kt = wave; kt < nkt; kt += 4) {
    int krow = kt * 16 + lm;
    int krc = krow < Skv ? krow : (Skv - 1);
    float mf = krow < Skv ? 1.f : 0.f;
    const float* kp = K + kbase + (size_t)krc * D;
    floatx8 f0 = *(const floatx8*)(kp + lk);
    floatx8 f1 = *(const floatx8*)(kp + lk + 8);
    floatx8 f2 = *(const floatx8*)(kp + 32 + lk);
    floatx8 f3 = *(const floatx8*)(kp + 32 + lk + 8);
    BF16x16 bk0, bk1;
    bk0.h[0] = cvt8(f0 * mf); bk0.h[1] = cvt8(f1 * mf);
    bk1.h[0] = cvt8(f2 * mf); bk1.h[1] = cvt8(f3 * mf);
    floatx8 acc = {};
    acc = wmma_bf16(aq0, bk0.v, acc);
    acc = wmma_bf16(aq1, bk1.v, acc);
    int col = kt * 16 + lm;
    int mr0 = (lane >> 4) * 8;
    bool valid = (col < Skv);
#pragma unroll
    for (int r = 0; r < 8; ++r)
      sS[(mr0 + r) * SKP + col] = valid ? acc[r] * scale : -__builtin_inff();
  }
  __syncthreads();

  // ---- softmax (8 threads per row) ----
  int srow = tid >> 3, ssub = tid & 7;
  {
    float mx = -__builtin_inff();
    for (int c = ssub; c < SKP; c += 8) mx = fmaxf(mx, sS[srow * SKP + c]);
    red[srow * 8 + ssub] = mx;
  }
  __syncthreads();
  if (tid < 16) {
    float mx = red[tid * 8];
#pragma unroll
    for (int i = 1; i < 8; ++i) mx = fmaxf(mx, red[tid * 8 + i]);
    rowv[tid] = mx;
  }
  __syncthreads();
  {
    float mx = rowv[srow], s = 0.f;
    for (int c = ssub; c < SKP; c += 8) {
      float e = __expf(sS[srow * SKP + c] - mx);
      sS[srow * SKP + c] = e;
      s += e;
    }
    __syncthreads();
    red[srow * 8 + ssub] = s;
  }
  __syncthreads();
  if (tid < 16) {
    float s = 0.f;
#pragma unroll
    for (int i = 0; i < 8; ++i) s += red[tid * 8 + i];
    rowv[tid] = 1.f / s;
  }
  __syncthreads();

  // ---- P x V: V^T fragments read contiguously from global; no barriers ----
  floatx8 oacc = {};
  int n = wave * 16 + lm;
  const float* vtp = VT + ((size_t)b * D + h * HD + n) * SKP;
  float rinv = rowv[lm];
  for (int kb = 0; kb < SKP; kb += 32) {
    floatx8 p0 = *(const floatx8*)&sS[lm * SKP + kb + lk];
    floatx8 p1 = *(const floatx8*)&sS[lm * SKP + kb + lk + 8];
    BF16x16 ap; ap.h[0] = cvt8(p0 * rinv); ap.h[1] = cvt8(p1 * rinv);
    floatx8 v0 = *(const floatx8*)(vtp + kb + lk);
    floatx8 v1 = *(const floatx8*)(vtp + kb + lk + 8);
    BF16x16 bv; bv.h[0] = cvt8(v0); bv.h[1] = cvt8(v1);
    oacc = wmma_bf16(ap.v, bv.v, oacc);
  }

  int mr0 = (lane >> 4) * 8;
  size_t obase = ((size_t)b * S + q0) * D + (size_t)h * HD + n;
#pragma unroll
  for (int r = 0; r < 8; ++r) {
    size_t idx = obase + (size_t)(mr0 + r) * D;
    float v = oacc[r];
    if (accum) v += O[idx];
    O[idx] = v;
  }
}

// ---------------------------------------------------------------------------
// GEGLU: g = a * gelu_tanh(gate)
// ---------------------------------------------------------------------------
__global__ __launch_bounds__(256) void k_geglu(const float* __restrict__ proj,
                                               float* __restrict__ g,
                                               int rows, int DFF) {
  size_t i = (size_t)blockIdx.x * blockDim.x + threadIdx.x;
  size_t total = (size_t)rows * DFF;
  if (i >= total) return;
  size_t r = i / DFF, c = i % DFF;
  float a  = proj[r * (size_t)(2 * DFF) + c];
  float gt = proj[r * (size_t)(2 * DFF) + DFF + c];
  float t = tanhf(0.7978845608028654f * (gt + 0.044715f * gt * gt * gt));
  g[i] = a * 0.5f * gt * (1.f + t);
}

// ---------------------------------------------------------------------------
// Host orchestration
// ---------------------------------------------------------------------------
static inline void gemm(hipStream_t st, const float* A, const __bf16* Wt,
                        const float* bias, const float* resid, float* C,
                        int M, int N, int K) {
  dim3 grid(N / BN, (M + BM - 1) / BM);
  k_gemm<<<grid, 256, 0, st>>>(A, Wt, bias, resid, C, M, N, K);
}

static inline void attn(hipStream_t st, const float* Q, const float* K,
                        const float* VT, float* O, int B, int S, int H,
                        int Skv, float scale, int accum) {
  int SKP = (Skv + 31) & ~31;
  size_t shmem = (size_t)16 * SKP * 4 + 16 * 64 * 2 + 16 * 8 * 4 + 16 * 4;
  dim3 grid(S / 16, H, B);
  k_attn16<<<grid, 128, shmem, st>>>(Q, K, VT, O, Skv, scale, accum);
}

extern "C" void kernel_launch(void* const* d_in, const int* in_sizes, int n_in,
                              void* d_out, int out_size, void* d_ws, size_t ws_size,
                              hipStream_t stream) {
  (void)in_sizes; (void)n_in; (void)out_size; (void)ws_size;
  const int B = 4, S = 1024, D = 1024, H = 16, SE = 77, DE = 768, DFF = 4096;
  const int rows = B * S;          // 4096
  const int encRows = B * SE;      // 308
  const int SEP = (SE + 31) & ~31; // 96
  const size_t BSD = (size_t)rows * D;
  const float scale = 0.125f;

  const float* h1  = (const float*)d_in[0];
  const float* h2  = (const float*)d_in[1];
  const float* enc = (const float*)d_in[2];
  const float* wq  = (const float*)d_in[3];
  const float* wk  = (const float*)d_in[4];
  const float* wv  = (const float*)d_in[5];
  const float* wqc = (const float*)d_in[6];
  const float* wo  = (const float*)d_in[7];
  const float* bo  = (const float*)d_in[8];
  const float* ln1g = (const float*)d_in[9];
  const float* ln1b = (const float*)d_in[10];
  const float* ln2g = (const float*)d_in[11];
  const float* ln2b = (const float*)d_in[12];
  const float* w2q = (const float*)d_in[13];
  const float* w2k = (const float*)d_in[14];
  const float* w2v = (const float*)d_in[15];
  const float* w2o = (const float*)d_in[16];
  const float* b2o = (const float*)d_in[17];
  const float* wff1 = (const float*)d_in[18];
  const float* bff1 = (const float*)d_in[19];
  const float* wff2 = (const float*)d_in[20];
  const float* bff2 = (const float*)d_in[21];

  // ---- workspace carve-out ----
  size_t off = 0;
  auto alloc = [&](size_t bytes) -> char* {
    char* p = (char*)d_ws + off;
    off += (bytes + 255) & ~(size_t)255;
    return p;
  };
  // transpose+convert a (2,K,N) weight -> two (N,K) bf16 slices
  auto cvtT = [&](const float* src, int K, int N) -> __bf16* {
    __bf16* dst = (__bf16*)alloc((size_t)2 * K * N * sizeof(__bf16));
    dim3 g(N / 32, K / 32);
    k_wtrans<<<g, 256, 0, stream>>>(src,                 dst,                 K, N);
    k_wtrans<<<g, 256, 0, stream>>>(src + (size_t)K * N, dst + (size_t)K * N, K, N);
    return dst;
  };

  __bf16* wq_t  = cvtT(wq,  D, D);
  __bf16* wk_t  = cvtT(wk,  D, D);
  __bf16* wv_t  = cvtT(wv,  D, D);
  __bf16* wqc_t = cvtT(wqc, D, D);
  __bf16* wo_t  = cvtT(wo,  D, D);
  __bf16* w2q_t = cvtT(w2q, D, D);
  __bf16* w2o_t = cvtT(w2o, D, D);
  __bf16* w2k_t = cvtT(w2k, DE, D);
  __bf16* w2v_t = cvtT(w2v, DE, D);
  __bf16* wff1_t = cvtT(wff1, D, 2 * DFF);
  __bf16* wff2_t = cvtT(wff2, DFF, D);

  float* n1 = (float*)alloc(BSD * 4);
  float* n2 = (float*)alloc(BSD * 4);
  float* qkv = (float*)alloc(8 * BSD * 4);
  float* q1 = qkv + 0 * BSD, *k1 = qkv + 1 * BSD, *v1 = qkv + 2 * BSD, *qc1 = qkv + 3 * BSD;
  float* q2 = qkv + 4 * BSD, *k2 = qkv + 5 * BSD, *v2 = qkv + 6 * BSD, *qc2 = qkv + 7 * BSD;
  float* aacc = (float*)alloc(2 * BSD * 4);       // a1,a2; later geglu out
  float* a1 = aacc, *a2 = aacc + BSD;
  float* vt = (float*)alloc(2 * BSD * 4);         // v1t, v2t  (B,1024,1024)
  float* v1t = vt, *v2t = vt + BSD;
  float* encb = (float*)alloc(4 * (size_t)encRows * D * 4);
  float* ek1 = encb, *ev1 = ek1 + (size_t)encRows * D,
       * ek2 = ev1 + (size_t)encRows * D, *ev2 = ek2 + (size_t)encRows * D;
  float* evt = (float*)alloc(2 * (size_t)B * D * SEP * 4);   // (B,1024,96) x2
  float* ev1t = evt, *ev2t = evt + (size_t)B * D * SEP;
  // overlays (regions dead by the time these are used)
  float* ffproj = qkv;    // rows x 8192
  float* gbuf   = aacc;   // rows x 4096

  float* H1 = (float*)d_out;
  float* H2 = (float*)d_out + BSD;

  // ================= norm1 =================
  k_layernorm<<<rows, 256, 0, stream>>>(h1, ln1g,     ln1b,     n1, D);
  k_layernorm<<<rows, 256, 0, stream>>>(h2, ln1g + D, ln1b + D, n2, D);

  // ================= self qkv projections =================
  const size_t DD = (size_t)D * D;
  gemm(stream, n1, wq_t,       nullptr, nullptr, q1,  rows, D, D);
  gemm(stream, n1, wk_t,       nullptr, nullptr, k1,  rows, D, D);
  gemm(stream, n1, wv_t,       nullptr, nullptr, v1,  rows, D, D);
  gemm(stream, n1, wqc_t,      nullptr, nullptr, qc1, rows, D, D);
  gemm(stream, n2, wq_t + DD,  nullptr, nullptr, q2,  rows, D, D);
  gemm(stream, n2, wk_t + DD,  nullptr, nullptr, k2,  rows, D, D);
  gemm(stream, n2, wv_t + DD,  nullptr, nullptr, v2,  rows, D, D);
  gemm(stream, n2, wqc_t + DD, nullptr, nullptr, qc2, rows, D, D);

  // transpose V for the PV stage
  {
    dim3 g(S / 32, D / 32, B);
    k_vtrans<<<g, 256, 0, stream>>>(v1, v1t, S, S);
    k_vtrans<<<g, 256, 0, stream>>>(v2, v2t, S, S);
  }

  // ================= dual-stream attention =================
  attn(stream, q1,  k1, v1t, a1, B, S, H, S, scale, 0);
  attn(stream, qc1, k2, v2t, a1, B, S, H, S, scale, 1);
  attn(stream, q2,  k2, v2t, a2, B, S, H, S, scale, 0);
  attn(stream, qc2, k1, v1t, a2, B, S, H, S, scale, 1);

  gemm(stream, a1, wo_t,      bo,     h1, H1, rows, D, D);
  gemm(stream, a2, wo_t + DD, bo + D, h2, H2, rows, D, D);

  // ================= norm2 =================
  k_layernorm<<<rows, 256, 0, stream>>>(H1, ln2g,     ln2b,     n1, D);
  k_layernorm<<<rows, 256, 0, stream>>>(H2, ln2g + D, ln2b + D, n2, D);

  // ================= cross attention to encoder =================
  const size_t DED = (size_t)DE * D;
  gemm(stream, n1,  w2q_t,       nullptr, nullptr, q1,  rows, D, D);
  gemm(stream, n2,  w2q_t + DD,  nullptr, nullptr, q2,  rows, D, D);
  gemm(stream, enc, w2k_t,       nullptr, nullptr, ek1, encRows, D, DE);
  gemm(stream, enc, w2v_t,       nullptr, nullptr, ev1, encRows, D, DE);
  gemm(stream, enc, w2k_t + DED, nullptr, nullptr, ek2, encRows, D, DE);
  gemm(stream, enc, w2v_t + DED, nullptr, nullptr, ev2, encRows, D, DE);
  {
    dim3 g(SEP / 32, D / 32, B);
    k_vtrans<<<g, 256, 0, stream>>>(ev1, ev1t, SE, SEP);
    k_vtrans<<<g, 256, 0, stream>>>(ev2, ev2t, SE, SEP);
  }
  attn(stream, q1, ek1, ev1t, a1, B, S, H, SE, scale, 0);
  attn(stream, q2, ek2, ev2t, a2, B, S, H, SE, scale, 0);

  gemm(stream, a1, w2o_t,      b2o,     H1, H1, rows, D, D);
  gemm(stream, a2, w2o_t + DD, b2o + D, H2, H2, rows, D, D);

  // ================= GEGLU feed-forward =================
  {
    size_t total = (size_t)rows * DFF;
    int gblocks = (int)((total + 255) / 256);
    gemm(stream, n1, wff1_t, bff1, nullptr, ffproj, rows, 2 * DFF, D);
    k_geglu<<<gblocks, 256, 0, stream>>>(ffproj, gbuf, rows, DFF);
    gemm(stream, gbuf, wff2_t, bff2, H1, H1, rows, D, DFF);

    gemm(stream, n2, wff1_t + (size_t)D * 2 * DFF, bff1 + 2 * DFF, nullptr, ffproj, rows, 2 * DFF, D);
    k_geglu<<<gblocks, 256, 0, stream>>>(ffproj, gbuf, rows, DFF);
    gemm(stream, gbuf, wff2_t + (size_t)DFF * D, bff2 + D, H2, H2, rows, D, DFF);
  }
}
